// IPAttnProcessor_75634374083341
// MI455X (gfx1250) — matrix operand order repaired and checked by reading, not compiled, not run on hardware
//
#include <hip/hip_runtime.h>
#include <hip/hip_bf16.h>

// ---------------- types ----------------
typedef __attribute__((ext_vector_type(16))) __bf16 v16bf;
typedef __attribute__((ext_vector_type(8)))  __bf16 v8bf;
typedef __attribute__((ext_vector_type(8)))  float  v8f;

#define EPSF 1e-6f

static constexpr int LIMG = 2048;
static constexpr int LIP  = 512;
static constexpr int D    = 3072;
static constexpr int DIP  = 1280;
static constexpr int H    = 24;
static constexpr int HD   = 128;
static constexpr int LK   = 2560;   // 2048 + 512

// workspace layout (bytes, 256-aligned regions)
static constexpr size_t OFF_SHIFT = 0;                                  // 1280 f32
static constexpr size_t OFF_SCALE = 5120;                               // 1280 f32
static constexpr size_t OFF_IPN   = 10240;                              // 512*1280 bf16
static constexpr size_t OFF_QBF   = OFF_IPN  + (size_t)LIP * DIP * 2;   // 24*2048*128 bf16
static constexpr size_t OFF_KALL  = OFF_QBF  + (size_t)H * LIMG * HD * 2; // 24*2560*128 bf16
static constexpr size_t OFF_VT    = OFF_KALL + (size_t)H * LK * HD * 2;   // 24*128*2560 bf16

// ---------------- helpers ----------------
__device__ __forceinline__ v8f wmma_bf16(v16bf a, v16bf b, v8f c) {
  return __builtin_amdgcn_wmma_f32_16x16x32_bf16(false, a, false, b, (short)0, c,
                                                 false, false);
}

// A-fragment: two contiguous 8-element bf16 runs
__device__ __forceinline__ v16bf make_a(const __bf16* p0, const __bf16* p1) {
  v8bf lo = *(const v8bf*)p0;
  v8bf hi = *(const v8bf*)p1;
  return __builtin_shufflevector(lo, hi, 0,1,2,3,4,5,6,7,8,9,10,11,12,13,14,15);
}

// B-fragment: 16 contiguous bf16 (32B aligned)
__device__ __forceinline__ v16bf load_b(const __bf16* p) {
  return *(const v16bf*)p;
}

// B-fragment from 16 contiguous f32 with inline conversion
__device__ __forceinline__ v16bf cvt_b_f32(const float* p) {
  v16bf r;
#pragma unroll
  for (int i = 0; i < 16; ++i) r[i] = (__bf16)p[i];
  return r;
}

__device__ __forceinline__ v8f zero8() {
  v8f z = {0.f, 0.f, 0.f, 0.f, 0.f, 0.f, 0.f, 0.f};
  return z;
}

// ---------------- kernel 1: e = silu(t_emb) @ w_ada^T + b_ada ----------------
__global__ void __launch_bounds__(256) ada_kernel(const float* __restrict__ t_emb,
                                                  const float* __restrict__ w_ada,
                                                  const float* __restrict__ b_ada,
                                                  float* __restrict__ shift,
                                                  float* __restrict__ scale) {
  int o = blockIdx.x * 256 + threadIdx.x;
  if (o >= 2 * DIP) return;
  const float* w = w_ada + (size_t)o * DIP;
  float acc = b_ada[o];
  for (int i = 0; i < DIP; ++i) {
    float t = t_emb[i];
    float s = t / (1.f + __expf(-t));   // silu
    acc = fmaf(s, w[i], acc);
  }
  if (o < DIP) shift[o] = acc;
  else         scale[o - DIP] = acc;
}

// ---------------- kernel 2: AdaLayerNorm of ip_hidden_states -> bf16 ----------
__global__ void __launch_bounds__(256) ipnorm_kernel(const float* __restrict__ x,
                                                     const float* __restrict__ shift,
                                                     const float* __restrict__ scale,
                                                     __bf16* __restrict__ out) {
  int l = blockIdx.x;
  const float* row = x + (size_t)l * DIP;
  float s = 0.f, s2 = 0.f;
  for (int i = threadIdx.x; i < DIP; i += 256) {
    float v = row[i];
    s += v; s2 += v * v;
  }
#pragma unroll
  for (int m = 1; m < 32; m <<= 1) {
    s  += __shfl_xor(s,  m, 32);
    s2 += __shfl_xor(s2, m, 32);
  }
  __shared__ float ss[8], ss2[8];
  int w = threadIdx.x >> 5;
  if ((threadIdx.x & 31) == 0) { ss[w] = s; ss2[w] = s2; }
  __syncthreads();
  s = 0.f; s2 = 0.f;
#pragma unroll
  for (int i = 0; i < 8; ++i) { s += ss[i]; s2 += ss2[i]; }
  float mu  = s * (1.f / DIP);
  float var = s2 * (1.f / DIP) - mu * mu;
  float inv = rsqrtf(var + EPSF);
  __bf16* orow = out + (size_t)l * DIP;
  for (int i = threadIdx.x; i < DIP; i += 256) {
    float xn = (row[i] - mu) * inv;
    orow[i] = (__bf16)(xn * (1.f + scale[i]) + shift[i]);
  }
}

// ---------------- kernel 3: pack img q/k/v (RMSNorm heads, bf16, V^T) ---------
__global__ void __launch_bounds__(256) qkvpack_kernel(const float* __restrict__ q_in,
                                                      const float* __restrict__ k_in,
                                                      const float* __restrict__ v_in,
                                                      const float* __restrict__ g_q,
                                                      const float* __restrict__ g_k,
                                                      __bf16* __restrict__ q_bf,
                                                      __bf16* __restrict__ k_all,
                                                      __bf16* __restrict__ v_t) {
  int wid  = blockIdx.x * 8 + (threadIdx.x >> 5);  // one wave per (l,h)
  int lane = threadIdx.x & 31;
  int h = wid % H;
  int l = wid / H;
  size_t base = (size_t)l * D + (size_t)h * HD + (size_t)lane * 4;
  float4 qv = *(const float4*)(q_in + base);
  float4 kv = *(const float4*)(k_in + base);
  float4 vv = *(const float4*)(v_in + base);
  float qs = qv.x * qv.x + qv.y * qv.y + qv.z * qv.z + qv.w * qv.w;
  float ks = kv.x * kv.x + kv.y * kv.y + kv.z * kv.z + kv.w * kv.w;
#pragma unroll
  for (int m = 1; m < 32; m <<= 1) {
    qs += __shfl_xor(qs, m, 32);
    ks += __shfl_xor(ks, m, 32);
  }
  float qr = rsqrtf(qs * (1.f / HD) + EPSF) * 0.08838834764831845f; // fold 1/sqrt(128)
  float kr = rsqrtf(ks * (1.f / HD) + EPSF);
  const float* qa = (const float*)&qv;
  const float* ka = (const float*)&kv;
  const float* va = (const float*)&vv;
#pragma unroll
  for (int j = 0; j < 4; ++j) {
    int d = lane * 4 + j;
    q_bf [((size_t)h * LIMG + l) * HD + d] = (__bf16)(qa[j] * qr * g_q[d]);
    k_all[((size_t)h * LK   + l) * HD + d] = (__bf16)(ka[j] * kr * g_k[d]);
    v_t  [((size_t)h * HD   + d) * LK + l] = (__bf16)(va[j]);
  }
}

// ---------------- kernel 4: ip projections via WMMA + RMSNorm + scatter -------
// one wave computes a 16-row x 128-col (one head) strip of BOTH ip_key & ip_value
__global__ void __launch_bounds__(256) ipproj_kernel(const __bf16* __restrict__ ipn,
                                                     const float* __restrict__ wk,
                                                     const float* __restrict__ wv,
                                                     const float* __restrict__ g_ipk,
                                                     __bf16* __restrict__ k_all,
                                                     __bf16* __restrict__ v_t) {
  int wid  = blockIdx.x * 8 + (threadIdx.x >> 5);
  int lane = threadIdx.x & 31;
  int h  = wid >> 5;         // 24 heads
  int rb = wid & 31;         // 32 row blocks of 16 (512 rows)
  int lbase = rb * 16;
  int hi = lane >> 4, lo = lane & 15;

  v8f accK[8], accV[8];
#pragma unroll
  for (int c = 0; c < 8; ++c) { accK[c] = zero8(); accV[c] = zero8(); }

  const __bf16* arow = ipn + (size_t)(lbase + lo) * DIP;
  for (int ib = 0; ib < DIP; ib += 32) {
    v16bf a = make_a(arow + ib + hi * 16, arow + ib + hi * 16 + 8);
    int ioff = ib + hi * 16;
#pragma unroll
    for (int c = 0; c < 8; ++c) {
      int o = h * HD + c * 16 + lo;                      // output column (row of w)
      v16bf bk = cvt_b_f32(wk + (size_t)o * DIP + ioff);
      v16bf bv = cvt_b_f32(wv + (size_t)o * DIP + ioff);
      accK[c] = wmma_bf16(a, bk, accK[c]);
      accV[c] = wmma_bf16(a, bv, accV[c]);
    }
  }

  // per-row RMS over 128 dims of ip_key (rows live across the 16-lane half)
  v8f sq;
#pragma unroll
  for (int r = 0; r < 8; ++r) {
    float t = 0.f;
#pragma unroll
    for (int c = 0; c < 8; ++c) t = fmaf(accK[c][r], accK[c][r], t);
    sq[r] = t;
  }
#pragma unroll
  for (int m = 1; m < 16; m <<= 1)
#pragma unroll
    for (int r = 0; r < 8; ++r) sq[r] += __shfl_xor(sq[r], m, 32);
  v8f rs;
#pragma unroll
  for (int r = 0; r < 8; ++r) rs[r] = rsqrtf(sq[r] * (1.f / HD) + EPSF);

#pragma unroll
  for (int c = 0; c < 8; ++c) {
    int dd = c * 16 + lo;
    float g = g_ipk[dd];
#pragma unroll
    for (int r = 0; r < 8; ++r) {
      int key = 2048 + lbase + r + 8 * hi;
      k_all[((size_t)h * LK + key) * HD + dd] = (__bf16)(accK[c][r] * rs[r] * g);
      v_t  [((size_t)h * HD + dd) * LK + key] = (__bf16)(accV[c][r]);
    }
  }
}

// ---------------- kernel 5: flash attention, one wave per 16-query tile -------
__global__ void __launch_bounds__(256) attn_kernel(const __bf16* __restrict__ q_bf,
                                                   const __bf16* __restrict__ k_all,
                                                   const __bf16* __restrict__ v_t,
                                                   float* __restrict__ out) {
  __shared__ __align__(32) __bf16 pbuf[8][16 * 32];
  int wslot = threadIdx.x >> 5;
  int wid   = blockIdx.x * 8 + wslot;
  int lane  = threadIdx.x & 31;
  int h  = wid >> 7;         // 24 heads
  int qt = wid & 127;        // 128 query tiles
  int qbase = qt * 16;
  int hi = lane >> 4, lo = lane & 15;

  // Q A-fragments (16 x 128 split into 4 K=32 chunks)
  const __bf16* qrow = q_bf + ((size_t)h * LIMG + qbase + lo) * HD;
  v16bf qa[4];
#pragma unroll
  for (int c = 0; c < 4; ++c)
    qa[c] = make_a(qrow + c * 32 + hi * 16, qrow + c * 32 + hi * 16 + 8);

  v8f o_acc[8];
#pragma unroll
  for (int c = 0; c < 8; ++c) o_acc[c] = zero8();
  v8f mrow, lrow;
#pragma unroll
  for (int r = 0; r < 8; ++r) { mrow[r] = -1e30f; lrow[r] = 0.f; }

  const __bf16* kb = k_all + (size_t)h * LK * HD;
  const __bf16* vb = v_t   + (size_t)h * HD * LK;
  __bf16* myp = &pbuf[wslot][0];

  for (int j = 0; j < LK; j += 32) {
    // S tile: 16 queries x 32 keys
    v8f s0 = zero8(), s1 = zero8();
#pragma unroll
    for (int c = 0; c < 4; ++c) {
      v16bf b0 = load_b(kb + (size_t)(j + lo)      * HD + c * 32 + hi * 16);
      v16bf b1 = load_b(kb + (size_t)(j + 16 + lo) * HD + c * 32 + hi * 16);
      s0 = wmma_bf16(qa[c], b0, s0);
      s1 = wmma_bf16(qa[c], b1, s1);
    }

    // online softmax (rows span 16-lane halves; xor masks < 16 stay in-half)
    v8f mx;
#pragma unroll
    for (int r = 0; r < 8; ++r) mx[r] = fmaxf(s0[r], s1[r]);
#pragma unroll
    for (int m = 1; m < 16; m <<= 1)
#pragma unroll
      for (int r = 0; r < 8; ++r) mx[r] = fmaxf(mx[r], __shfl_xor(mx[r], m, 32));

    v8f alpha, p0, p1, rsum;
#pragma unroll
    for (int r = 0; r < 8; ++r) {
      float mn = fmaxf(mrow[r], mx[r]);
      alpha[r] = __expf(mrow[r] - mn);
      p0[r] = __expf(s0[r] - mn);
      p1[r] = __expf(s1[r] - mn);
      rsum[r] = p0[r] + p1[r];
      mrow[r] = mn;
    }
#pragma unroll
    for (int m = 1; m < 16; m <<= 1)
#pragma unroll
      for (int r = 0; r < 8; ++r) rsum[r] += __shfl_xor(rsum[r], m, 32);
#pragma unroll
    for (int r = 0; r < 8; ++r) lrow[r] = lrow[r] * alpha[r] + rsum[r];
#pragma unroll
    for (int c = 0; c < 8; ++c)
#pragma unroll
      for (int r = 0; r < 8; ++r) o_acc[c][r] *= alpha[r];

    // stage P (C-layout) -> LDS -> A-layout bf16 fragment
#pragma unroll
    for (int r = 0; r < 8; ++r) {
      int row = r + 8 * hi;
      myp[row * 32 + lo]      = (__bf16)p0[r];
      myp[row * 32 + 16 + lo] = (__bf16)p1[r];
    }
    asm volatile("s_wait_dscnt 0" ::: "memory");  // cross-lane LDS RAW within wave
    v16bf pa = make_a(myp + lo * 32 + hi * 16, myp + lo * 32 + hi * 16 + 8);

    // O += P @ V  (V^T layout -> contiguous B fragments)
#pragma unroll
    for (int c = 0; c < 8; ++c) {
      v16bf bv = load_b(vb + (size_t)(c * 16 + lo) * LK + j + hi * 16);
      o_acc[c] = wmma_bf16(pa, bv, o_acc[c]);
    }
  }

  // epilogue: normalize and write [l, h*128 + d] f32
#pragma unroll
  for (int r = 0; r < 8; ++r) {
    float il = 1.f / lrow[r];
    size_t orow = (size_t)(qbase + r + 8 * hi) * D + (size_t)h * HD;
#pragma unroll
    for (int c = 0; c < 8; ++c)
      out[orow + c * 16 + lo] = o_acc[c][r] * il;
  }
}

// ---------------- launch ----------------
extern "C" void kernel_launch(void* const* d_in, const int* in_sizes, int n_in,
                              void* d_out, int out_size, void* d_ws, size_t ws_size,
                              hipStream_t stream) {
  (void)in_sizes; (void)n_in; (void)out_size; (void)ws_size;
  const float* ip_hidden = (const float*)d_in[0];
  const float* img_q     = (const float*)d_in[1];
  const float* img_k     = (const float*)d_in[2];
  const float* img_v     = (const float*)d_in[3];
  const float* t_emb     = (const float*)d_in[4];
  const float* w_ada     = (const float*)d_in[5];
  const float* b_ada     = (const float*)d_in[6];
  const float* w_k_ip    = (const float*)d_in[7];
  const float* w_v_ip    = (const float*)d_in[8];
  const float* g_q       = (const float*)d_in[9];
  const float* g_k       = (const float*)d_in[10];
  const float* g_ipk     = (const float*)d_in[11];

  char* ws = (char*)d_ws;
  float*  shift = (float*)(ws + OFF_SHIFT);
  float*  scale = (float*)(ws + OFF_SCALE);
  __bf16* ipn   = (__bf16*)(ws + OFF_IPN);
  __bf16* q_bf  = (__bf16*)(ws + OFF_QBF);
  __bf16* k_all = (__bf16*)(ws + OFF_KALL);
  __bf16* v_t   = (__bf16*)(ws + OFF_VT);
  float*  outp  = (float*)d_out;

  ada_kernel   <<<(2 * DIP + 255) / 256, 256, 0, stream>>>(t_emb, w_ada, b_ada, shift, scale);
  ipnorm_kernel<<<LIP, 256, 0, stream>>>(ip_hidden, shift, scale, ipn);
  qkvpack_kernel<<<(H * LIMG) / 8, 256, 0, stream>>>(img_q, img_k, img_v, g_q, g_k,
                                                     q_bf, k_all, v_t);
  ipproj_kernel<<<(H * 32) / 8, 256, 0, stream>>>(ipn, w_k_ip, w_v_ip, g_ipk, k_all, v_t);
  attn_kernel  <<<(H * 128) / 8, 256, 0, stream>>>(q_bf, k_all, v_t, outp);
}
